// AttentionBlock_27960237097380
// MI455X (gfx1250) — matrix operand order repaired and checked by reading, not compiled, not run on hardware
//
#include <hip/hip_runtime.h>
#include <hip/hip_bf16.h>

typedef __attribute__((ext_vector_type(16))) __bf16 v16bf;
typedef __attribute__((ext_vector_type(8)))  float  v8f;

#define CC   512
#define NN   4096
#define BB   8
#define OC3  1536

// ---------------------------------------------------------------------------
// f32 -> bf16 convert
// ---------------------------------------------------------------------------
__global__ void cvt_f32_bf16(const float* __restrict__ in, __bf16* __restrict__ out, int n) {
    int i = blockIdx.x * 256 + threadIdx.x;
    if (i < n) out[i] = (__bf16)in[i];
}

// ---------------------------------------------------------------------------
// GroupNorm: x[b][c][n] f32 -> h_t[b][n][c] bf16 (transposed layout)
// grid: (32 groups, 8 batch), 256 threads. 16 channels x 4096 px per block.
// ---------------------------------------------------------------------------
__global__ __launch_bounds__(256) void groupnorm_t(
    const float* __restrict__ x, const float* __restrict__ gw,
    const float* __restrict__ gb, __bf16* __restrict__ h_t) {
    const int g = blockIdx.x, b = blockIdx.y, tid = threadIdx.x;
    const float* xg = x + ((size_t)b * CC + g * 16) * NN;

    __shared__ float rsum[256], rsq[256];
    __shared__ float sa[16], sb[16];
    __shared__ __attribute__((aligned(16))) __bf16 tile[16 * 256];

    float s = 0.f, s2 = 0.f;
    for (int i = 0; i < 256; ++i) {
        int idx = i * 256 + tid;
        float v = xg[(idx >> 12) * NN + (idx & (NN - 1))];
        s += v; s2 += v * v;
    }
    rsum[tid] = s; rsq[tid] = s2;
    __syncthreads();
    for (int st = 128; st > 0; st >>= 1) {
        if (tid < st) { rsum[tid] += rsum[tid + st]; rsq[tid] += rsq[tid + st]; }
        __syncthreads();
    }
    if (tid < 16) {
        float mean = rsum[0] * (1.f / 65536.f);
        float var  = rsq[0] * (1.f / 65536.f) - mean * mean;
        float rstd = rsqrtf(var + 1e-5f);
        int gc = g * 16 + tid;
        float a = gw[gc] * rstd;
        sa[tid] = a;
        sb[tid] = gb[gc] - mean * a;
    }
    __syncthreads();

    // transpose-write via LDS: 16c x 256n tiles
    for (int nt = 0; nt < NN; nt += 256) {
        int c = tid >> 4, j0 = (tid & 15) * 16;
        const float* row = xg + c * NN + nt + j0;
        float a = sa[c], bb = sb[c];
        #pragma unroll
        for (int u = 0; u < 16; ++u)
            tile[c * 256 + j0 + u] = (__bf16)(row[u] * a + bb);
        __syncthreads();
        int cw = tid & 15, nb = tid >> 4;
        #pragma unroll
        for (int u = 0; u < 16; ++u) {
            int nl = nb * 16 + u;
            h_t[((size_t)b * NN + nt + nl) * CC + g * 16 + cw] = tile[cw * 256 + nl];
        }
        __syncthreads();
    }
}

// ---------------------------------------------------------------------------
// wave reductions (wave32)
// ---------------------------------------------------------------------------
__device__ inline float wave_max(float v) {
    for (int o = 16; o; o >>= 1) v = fmaxf(v, __shfl_xor(v, o, 32));
    return v;
}
__device__ inline float wave_sum(float v) {
    for (int o = 16; o; o >>= 1) v += __shfl_xor(v, o, 32);
    return v;
}

// ---------------------------------------------------------------------------
// softmax over n for k rows, in place. grid: (64 rowgroups, 8 batch), 8 waves,
// one row per wave.
// ---------------------------------------------------------------------------
__global__ __launch_bounds__(256) void softmax_k(__bf16* __restrict__ qkv) {
    const int lane = threadIdx.x & 31, wave = threadIdx.x >> 5;
    const int row = blockIdx.x * 8 + wave, b = blockIdx.y;
    __bf16* p = qkv + ((size_t)b * OC3 + CC + row) * NN;

    float m = -3.4e38f;
    for (int i = 0; i < 128; ++i) m = fmaxf(m, (float)p[i * 32 + lane]);
    m = wave_max(m);
    float s = 0.f;
    for (int i = 0; i < 128; ++i) s += __expf((float)p[i * 32 + lane] - m);
    s = wave_sum(s);
    float inv = 1.f / s;
    for (int i = 0; i < 128; ++i) {
        int n = i * 32 + lane;
        p[n] = (__bf16)(__expf((float)p[n] - m) * inv);
    }
}

// ---------------------------------------------------------------------------
// softmax over n for q rows, scaled by C^-0.5, transposed out: q_t[b][n][c].
// grid: (32 rowgroups of 16, 8 batch), 256 threads (8 waves, 2 rows each).
// ---------------------------------------------------------------------------
__global__ __launch_bounds__(256) void softmax_q_t(
    const __bf16* __restrict__ qkv, __bf16* __restrict__ q_t) {
    const int lane = threadIdx.x & 31, wave = threadIdx.x >> 5;
    const int c0 = blockIdx.x * 16, b = blockIdx.y, tid = threadIdx.x;

    __shared__ float smax[16], sinv[16];
    __shared__ __attribute__((aligned(16))) __bf16 tile[16 * 256];

    const float scale = rsqrtf((float)CC);
    for (int rr = 0; rr < 2; ++rr) {
        int row = c0 + wave * 2 + rr;
        const __bf16* p = qkv + ((size_t)b * OC3 + row) * NN;
        float m = -3.4e38f;
        for (int i = 0; i < 128; ++i) m = fmaxf(m, (float)p[i * 32 + lane]);
        m = wave_max(m);
        float s = 0.f;
        for (int i = 0; i < 128; ++i) s += __expf((float)p[i * 32 + lane] - m);
        s = wave_sum(s);
        if (lane == 0) { smax[wave * 2 + rr] = m; sinv[wave * 2 + rr] = scale / s; }
    }
    __syncthreads();

    for (int nt = 0; nt < NN; nt += 256) {
        int c = tid >> 4, j0 = (tid & 15) * 16;
        const __bf16* row = qkv + ((size_t)b * OC3 + c0 + c) * NN + nt + j0;
        float m = smax[c], inv = sinv[c];
        #pragma unroll
        for (int u = 0; u < 16; ++u)
            tile[c * 256 + j0 + u] = (__bf16)(__expf((float)row[u] - m) * inv);
        __syncthreads();
        int cw = tid & 15, nb = tid >> 4;
        #pragma unroll
        for (int u = 0; u < 16; ++u) {
            int nl = nb * 16 + u;
            q_t[((size_t)b * NN + nt + nl) * CC + c0 + cw] = tile[cw * 256 + nl];
        }
        __syncthreads();
    }
}

// ---------------------------------------------------------------------------
// WMMA fragment loaders (16-bit A 16x32 / B 32x16; contiguous k = 16*(lane/16) + s in both)
// ---------------------------------------------------------------------------
#define BK 32
__device__ inline v16bf load_frag_a(const __bf16* sm, int m0, int lane) {
    const __bf16* row = sm + (m0 + (lane & 15)) * BK;
    int ko = (lane < 16) ? 0 : 16;
    v16bf f;
    #pragma unroll
    for (int j = 0; j < 8; ++j) {
        int k = ko + 2 * j;
        f[2 * j] = row[k]; f[2 * j + 1] = row[k + 1];
    }
    return f;
}
__device__ inline v16bf load_frag_b(const __bf16* sm, int n0, int lane) {
    const __bf16* row = sm + (n0 + (lane & 15)) * BK;
    int ko = (lane < 16) ? 0 : 16;
    v16bf f;
    #pragma unroll
    for (int j = 0; j < 8; ++j) {
        int k = ko + 2 * j;
        f[2 * j] = row[k]; f[2 * j + 1] = row[k + 1];
    }
    return f;
}

// ---------------------------------------------------------------------------
// bf16 GEMM: OUT[m,n] = sum_k A[m,k] * Bt[n,k]  (A:[M,K], Bt:[N,K], row-major)
// MODE 0: bf16 out + per-m bias; MODE 1: bf16 out; MODE 2: f32 out + bias + residual
// Block 256 thr (8 waves), tile 128x128, BK=32, wave = 32x64 (2x4 wmma 16x16)
// ---------------------------------------------------------------------------
template <int MODE>
__global__ __launch_bounds__(256) void gemm_bf16(
    const __bf16* __restrict__ A, const __bf16* __restrict__ Bt, void* __restrict__ OutV,
    const float* __restrict__ bias, const float* __restrict__ resid,
    int M, int N, int K, long asb, long bsb, long osb, long rsb) {
    const int tid = threadIdx.x, lane = tid & 31, wave = tid >> 5;
    const int wm = wave >> 1, wn = wave & 1;          // 4 x 2 waves
    const int bm = blockIdx.y * 128, bn = blockIdx.x * 128, z = blockIdx.z;

    A  += (size_t)z * asb;
    Bt += (size_t)z * bsb;

    __shared__ __attribute__((aligned(16))) __bf16 As[128 * BK];
    __shared__ __attribute__((aligned(16))) __bf16 Bs[128 * BK];

    v8f acc[2][4] = {};

    for (int kt = 0; kt < K; kt += BK) {
        // A and B tiles: each 128x32 bf16 = 512 16B quads -> 2 quads/thread each
        #pragma unroll
        for (int i = 0; i < 2; ++i) {
            int qid = tid + i * 256;
            int row = qid >> 2, q = qid & 3;
            *(uint4*)(As + row * BK + q * 8) =
                *(const uint4*)(A + (size_t)(bm + row) * K + kt + q * 8);
            *(uint4*)(Bs + row * BK + q * 8) =
                *(const uint4*)(Bt + (size_t)(bn + row) * K + kt + q * 8);
        }
        if (kt + BK < K) {
            __builtin_prefetch(A + (size_t)(bm + (tid >> 1)) * K + kt + BK, 0, 3);
            __builtin_prefetch(Bt + (size_t)(bn + (tid >> 1)) * K + kt + BK, 0, 3);
        }
        __syncthreads();

        v16bf a0 = load_frag_a(As, wm * 32,      lane);
        v16bf a1 = load_frag_a(As, wm * 32 + 16, lane);
        v16bf b0 = load_frag_b(Bs, wn * 64,      lane);
        v16bf b1 = load_frag_b(Bs, wn * 64 + 16, lane);
        v16bf b2 = load_frag_b(Bs, wn * 64 + 32, lane);
        v16bf b3 = load_frag_b(Bs, wn * 64 + 48, lane);

        acc[0][0] = __builtin_amdgcn_wmma_f32_16x16x32_bf16(false, a0, false, b0, (short)0, acc[0][0], false, false);
        acc[0][1] = __builtin_amdgcn_wmma_f32_16x16x32_bf16(false, a0, false, b1, (short)0, acc[0][1], false, false);
        acc[0][2] = __builtin_amdgcn_wmma_f32_16x16x32_bf16(false, a0, false, b2, (short)0, acc[0][2], false, false);
        acc[0][3] = __builtin_amdgcn_wmma_f32_16x16x32_bf16(false, a0, false, b3, (short)0, acc[0][3], false, false);
        acc[1][0] = __builtin_amdgcn_wmma_f32_16x16x32_bf16(false, a1, false, b0, (short)0, acc[1][0], false, false);
        acc[1][1] = __builtin_amdgcn_wmma_f32_16x16x32_bf16(false, a1, false, b1, (short)0, acc[1][1], false, false);
        acc[1][2] = __builtin_amdgcn_wmma_f32_16x16x32_bf16(false, a1, false, b2, (short)0, acc[1][2], false, false);
        acc[1][3] = __builtin_amdgcn_wmma_f32_16x16x32_bf16(false, a1, false, b3, (short)0, acc[1][3], false, false);
        __syncthreads();
    }

    const int hi8 = (lane & 16) ? 8 : 0;
    #pragma unroll
    for (int i = 0; i < 2; ++i)
        #pragma unroll
        for (int j = 0; j < 4; ++j)
            #pragma unroll
            for (int r = 0; r < 8; ++r) {
                int m = bm + wm * 32 + i * 16 + r + hi8;
                int n = bn + wn * 64 + j * 16 + (lane & 15);
                float v = acc[i][j][r];
                if (MODE == 0) {
                    ((__bf16*)OutV)[(size_t)z * osb + (size_t)m * N + n] = (__bf16)(v + bias[m]);
                } else if (MODE == 1) {
                    ((__bf16*)OutV)[(size_t)z * osb + (size_t)m * N + n] = (__bf16)v;
                } else {
                    size_t o = (size_t)z * osb + (size_t)m * N + n;
                    ((float*)OutV)[o] = v + bias[m] + resid[(size_t)z * rsb + (size_t)m * N + n];
                }
            }
}

// ---------------------------------------------------------------------------
extern "C" void kernel_launch(void* const* d_in, const int* in_sizes, int n_in,
                              void* d_out, int out_size, void* d_ws, size_t ws_size,
                              hipStream_t stream) {
    const float* x      = (const float*)d_in[0];
    const float* norm_w = (const float*)d_in[1];
    const float* norm_b = (const float*)d_in[2];
    const float* qkv_w  = (const float*)d_in[3];
    const float* qkv_b  = (const float*)d_in[4];
    const float* proj_w = (const float*)d_in[5];
    const float* proj_b = (const float*)d_in[6];
    (void)in_sizes; (void)n_in; (void)out_size; (void)ws_size;

    char* ws = (char*)d_ws;
    size_t off = 0;
    auto alloc = [&](size_t bytes) { char* p = ws + off; off = (off + bytes + 255) & ~(size_t)255; return p; };
    __bf16* h_t     = (__bf16*)alloc((size_t)BB * NN * CC * 2);   // also reused as out2^T
    __bf16* qkv     = (__bf16*)alloc((size_t)BB * OC3 * NN * 2);
    __bf16* q_t     = (__bf16*)alloc((size_t)BB * NN * CC * 2);
    __bf16* ctx     = (__bf16*)alloc((size_t)BB * CC * CC * 2);
    __bf16* w_qkv   = (__bf16*)alloc((size_t)OC3 * CC * 2);
    __bf16* w_proj  = (__bf16*)alloc((size_t)CC * CC * 2);

    cvt_f32_bf16<<<(OC3 * CC + 255) / 256, 256, 0, stream>>>(qkv_w, w_qkv, OC3 * CC);
    cvt_f32_bf16<<<(CC * CC + 255) / 256, 256, 0, stream>>>(proj_w, w_proj, CC * CC);

    groupnorm_t<<<dim3(32, BB), 256, 0, stream>>>(x, norm_w, norm_b, h_t);

    // qkv = W_qkv @ h : M=1536, N=4096, K=512
    gemm_bf16<0><<<dim3(NN / 128, OC3 / 128, BB), 256, 0, stream>>>(
        w_qkv, h_t, qkv, qkv_b, nullptr, OC3, NN, CC,
        0L, (long)NN * CC, (long)OC3 * NN, 0L);

    softmax_k<<<dim3(CC / 8, BB), 256, 0, stream>>>(qkv);
    softmax_q_t<<<dim3(CC / 16, BB), 256, 0, stream>>>(qkv, q_t);

    // context[c,d] = sum_n k_s[c,n] v[d,n] : M=512, N=512, K=4096
    gemm_bf16<1><<<dim3(CC / 128, CC / 128, BB), 256, 0, stream>>>(
        qkv + (size_t)CC * NN, qkv + (size_t)2 * CC * NN, ctx, nullptr, nullptr,
        CC, CC, NN, (long)OC3 * NN, (long)OC3 * NN, (long)CC * CC, 0L);

    // out2^T[n,c] = sum_d q_t[n,d] ctx[c,d] : M=4096, N=512, K=512 (into h_t)
    gemm_bf16<1><<<dim3(CC / 128, NN / 128, BB), 256, 0, stream>>>(
        q_t, ctx, h_t, nullptr, nullptr,
        NN, CC, CC, (long)NN * CC, (long)CC * CC, (long)NN * CC, 0L);

    // y[o,n] = sum_c proj_w[o,c] out2^T[n,c] + proj_b[o] + x[o,n] : M=512, N=4096, K=512
    gemm_bf16<2><<<dim3(NN / 128, CC / 128, BB), 256, 0, stream>>>(
        w_proj, h_t, d_out, proj_b, x,
        CC, NN, CC, 0L, (long)NN * CC, (long)CC * NN, (long)CC * NN);
}
